// SpikingMLP_51719996178451
// MI455X (gfx1250) — compile-verified, matrix operand order checked by reading, not run on hardware
//
#include <hip/hip_runtime.h>

// ---------------------------------------------------------------------------
// SpikingMLP for MI455X (gfx1250, wave32, WMMA).
//   GEMM1 (x @ w_up^T) in bf16x3 split precision + fused LIF epilogue -> rate (bf16, exact)
//   GEMM2 (rate @ w_down^T) with bf16x2 split of w_down -> fp32 out
//   Deterministic two-pass column reduction for rate_per_unit.
// ---------------------------------------------------------------------------

typedef __bf16 bf16;
typedef __attribute__((ext_vector_type(8)))  bf16  v8bf;
typedef __attribute__((ext_vector_type(16))) bf16  v16bf;
typedef __attribute__((ext_vector_type(8)))  float v8f;

#define NTOK 4096   // B*S
#define DDIM 1024
#define FDIM 4096
#define BM   128
#define BN   128
#define BK   32
#define LDST 40     // padded LDS row stride (elements) -> conflict-free b128 reads

__device__ __forceinline__ v16bf make_frag(v8bf a, v8bf b) {
    return __builtin_shufflevector(a, b, 0,1,2,3,4,5,6,7,8,9,10,11,12,13,14,15);
}
#define FRAG(arr, buf, r, kb) \
    make_frag(*(const v8bf*)&arr[buf][r][kb], *(const v8bf*)&arr[buf][r][(kb) + 16])

#define WMMA_BF16(A, B, C) \
    __builtin_amdgcn_wmma_f32_16x16x32_bf16(false, (A), false, (B), (short)0, (C), false, false)

// --------------------------- fp32 -> (hi,lo) bf16 ---------------------------
__global__ void k_split(const float* __restrict__ in, bf16* __restrict__ hi,
                        bf16* __restrict__ lo, int n) {
    int i = blockIdx.x * blockDim.x + threadIdx.x;
    if (i < n) {
        float v = in[i];
        bf16  h = (bf16)v;
        hi[i] = h;
        lo[i] = (bf16)(v - (float)h);
    }
}

// --------------------- GEMM1 (bf16x3) + fused LIF epilogue -------------------
__global__ __launch_bounds__(256) void k_gemm1_lif(
    const bf16* __restrict__ xh, const bf16* __restrict__ xl,
    const bf16* __restrict__ wh, const bf16* __restrict__ wl,
    const float* __restrict__ beta, const int* __restrict__ Tp,
    bf16* __restrict__ rate)
{
    __shared__ __align__(16) bf16 sAh[2][BM][LDST];
    __shared__ __align__(16) bf16 sAl[2][BM][LDST];
    __shared__ __align__(16) bf16 sBh[2][BN][LDST];
    __shared__ __align__(16) bf16 sBl[2][BN][LDST];

    const int tid   = threadIdx.x;
    const int lane  = tid & 31;
    const int wave  = tid >> 5;
    const int waveM = wave & 3;          // 4 waves along M (tokens)
    const int waveN = wave >> 2;         // 2 waves along N (features)
    const int lrow  = lane & 15;
    const int khalf = (lane >> 4) * 8;   // per-lane K base / C-row offset

    const int n0 = blockIdx.y * BM;
    const int f0 = blockIdx.x * BN;

    const int ldrow = tid >> 1;          // 0..127
    const int ldcol = (tid & 1) * 16;    // 0 or 16

    const size_t aBase = (size_t)(n0 + ldrow) * DDIM + ldcol;
    const size_t bBase = (size_t)(f0 + ldrow) * DDIM + ldcol;

    v8f acc[2][4];
    #pragma unroll
    for (int i = 0; i < 2; ++i)
        #pragma unroll
        for (int j = 0; j < 4; ++j)
            acc[i][j] = (v8f){0.f,0.f,0.f,0.f,0.f,0.f,0.f,0.f};

    v8bf rAh0, rAh1, rAl0, rAl1, rBh0, rBh1, rBl0, rBl1;
    auto prefetch = [&](int ks) {
        const size_t ko = (size_t)ks * BK;
        rAh0 = *(const v8bf*)(xh + aBase + ko);
        rAh1 = *(const v8bf*)(xh + aBase + ko + 8);
        rAl0 = *(const v8bf*)(xl + aBase + ko);
        rAl1 = *(const v8bf*)(xl + aBase + ko + 8);
        rBh0 = *(const v8bf*)(wh + bBase + ko);
        rBh1 = *(const v8bf*)(wh + bBase + ko + 8);
        rBl0 = *(const v8bf*)(wl + bBase + ko);
        rBl1 = *(const v8bf*)(wl + bBase + ko + 8);
    };
    auto stage = [&](int buf) {
        *(v8bf*)&sAh[buf][ldrow][ldcol]     = rAh0;
        *(v8bf*)&sAh[buf][ldrow][ldcol + 8] = rAh1;
        *(v8bf*)&sAl[buf][ldrow][ldcol]     = rAl0;
        *(v8bf*)&sAl[buf][ldrow][ldcol + 8] = rAl1;
        *(v8bf*)&sBh[buf][ldrow][ldcol]     = rBh0;
        *(v8bf*)&sBh[buf][ldrow][ldcol + 8] = rBh1;
        *(v8bf*)&sBl[buf][ldrow][ldcol]     = rBl0;
        *(v8bf*)&sBl[buf][ldrow][ldcol + 8] = rBl1;
    };
    auto compute = [&](int buf) {
        v16bf ah[2], al[2];
        #pragma unroll
        for (int mi = 0; mi < 2; ++mi) {
            const int r = waveM * 32 + mi * 16 + lrow;
            ah[mi] = FRAG(sAh, buf, r, khalf);
            al[mi] = FRAG(sAl, buf, r, khalf);
        }
        #pragma unroll
        for (int ni = 0; ni < 4; ++ni) {
            const int r = waveN * 64 + ni * 16 + lrow;
            v16bf bh = FRAG(sBh, buf, r, khalf);
            v16bf bl = FRAG(sBl, buf, r, khalf);
            #pragma unroll
            for (int mi = 0; mi < 2; ++mi) {
                acc[mi][ni] = WMMA_BF16(ah[mi], bh, acc[mi][ni]);
                acc[mi][ni] = WMMA_BF16(ah[mi], bl, acc[mi][ni]);
                acc[mi][ni] = WMMA_BF16(al[mi], bh, acc[mi][ni]);
            }
        }
    };

    const int NK = DDIM / BK;            // 32
    prefetch(0);
    stage(0);
    __syncthreads();
    for (int ks = 0; ks < NK; ++ks) {
        const bool more = (ks + 1 < NK);
        if (more) prefetch(ks + 1);
        compute(ks & 1);
        if (more) {
            __syncthreads();
            stage((ks + 1) & 1);
            __syncthreads();
        }
    }

    // --------- fused LIF epilogue: h -> rate (exact multiples of 1/T) --------
    const int   T    = *Tp;
    const float invT = 1.0f / (float)T;
    #pragma unroll
    for (int ni = 0; ni < 4; ++ni) {
        const int   f   = f0 + waveN * 64 + ni * 16 + lrow;
        const float bta = beta[f];
        #pragma unroll
        for (int mi = 0; mi < 2; ++mi) {
            const int nb = n0 + waveM * 32 + mi * 16 + khalf;
            #pragma unroll
            for (int r = 0; r < 8; ++r) {
                const float h = acc[mi][ni][r];
                float v = 0.f, ss = 0.f;
                for (int t = 0; t < T; ++t) {
                    v = bta * v + h;
                    const float s = (v > 1.0f) ? 1.0f : 0.0f;  // THRESHOLD = 1
                    v -= s;
                    ss += s;
                }
                rate[(size_t)(nb + r) * FDIM + f] = (bf16)(ss * invT);
            }
        }
    }
}

// --------------------- GEMM2: out = rate @ w_down^T (bf16x2) -----------------
__global__ __launch_bounds__(256) void k_gemm2(
    const bf16* __restrict__ rt,
    const bf16* __restrict__ wdh, const bf16* __restrict__ wdl,
    float* __restrict__ out)
{
    __shared__ __align__(16) bf16 sA [2][BM][LDST];
    __shared__ __align__(16) bf16 sBh[2][BN][LDST];
    __shared__ __align__(16) bf16 sBl[2][BN][LDST];

    const int tid   = threadIdx.x;
    const int lane  = tid & 31;
    const int wave  = tid >> 5;
    const int waveM = wave & 3;
    const int waveN = wave >> 2;
    const int lrow  = lane & 15;
    const int khalf = (lane >> 4) * 8;

    const int n0 = blockIdx.y * BM;
    const int d0 = blockIdx.x * BN;

    const int ldrow = tid >> 1;
    const int ldcol = (tid & 1) * 16;

    const size_t aBase = (size_t)(n0 + ldrow) * FDIM + ldcol;
    const size_t bBase = (size_t)(d0 + ldrow) * FDIM + ldcol;

    v8f acc[2][4];
    #pragma unroll
    for (int i = 0; i < 2; ++i)
        #pragma unroll
        for (int j = 0; j < 4; ++j)
            acc[i][j] = (v8f){0.f,0.f,0.f,0.f,0.f,0.f,0.f,0.f};

    v8bf rA0, rA1, rBh0, rBh1, rBl0, rBl1;
    auto prefetch = [&](int ks) {
        const size_t ko = (size_t)ks * BK;
        rA0  = *(const v8bf*)(rt  + aBase + ko);
        rA1  = *(const v8bf*)(rt  + aBase + ko + 8);
        rBh0 = *(const v8bf*)(wdh + bBase + ko);
        rBh1 = *(const v8bf*)(wdh + bBase + ko + 8);
        rBl0 = *(const v8bf*)(wdl + bBase + ko);
        rBl1 = *(const v8bf*)(wdl + bBase + ko + 8);
    };
    auto stage = [&](int buf) {
        *(v8bf*)&sA [buf][ldrow][ldcol]     = rA0;
        *(v8bf*)&sA [buf][ldrow][ldcol + 8] = rA1;
        *(v8bf*)&sBh[buf][ldrow][ldcol]     = rBh0;
        *(v8bf*)&sBh[buf][ldrow][ldcol + 8] = rBh1;
        *(v8bf*)&sBl[buf][ldrow][ldcol]     = rBl0;
        *(v8bf*)&sBl[buf][ldrow][ldcol + 8] = rBl1;
    };
    auto compute = [&](int buf) {
        v16bf a[2];
        #pragma unroll
        for (int mi = 0; mi < 2; ++mi)
            a[mi] = FRAG(sA, buf, waveM * 32 + mi * 16 + lrow, khalf);
        #pragma unroll
        for (int ni = 0; ni < 4; ++ni) {
            const int r = waveN * 64 + ni * 16 + lrow;
            v16bf bh = FRAG(sBh, buf, r, khalf);
            v16bf bl = FRAG(sBl, buf, r, khalf);
            #pragma unroll
            for (int mi = 0; mi < 2; ++mi) {
                acc[mi][ni] = WMMA_BF16(a[mi], bh, acc[mi][ni]);
                acc[mi][ni] = WMMA_BF16(a[mi], bl, acc[mi][ni]);
            }
        }
    };

    const int NK = FDIM / BK;            // 128
    prefetch(0);
    stage(0);
    __syncthreads();
    for (int ks = 0; ks < NK; ++ks) {
        const bool more = (ks + 1 < NK);
        if (more) prefetch(ks + 1);
        compute(ks & 1);
        if (more) {
            __syncthreads();
            stage((ks + 1) & 1);
            __syncthreads();
        }
    }

    #pragma unroll
    for (int ni = 0; ni < 4; ++ni) {
        const int d = d0 + waveN * 64 + ni * 16 + lrow;
        #pragma unroll
        for (int mi = 0; mi < 2; ++mi) {
            const int nb = n0 + waveM * 32 + mi * 16 + khalf;
            #pragma unroll
            for (int r = 0; r < 8; ++r)
                out[(size_t)(nb + r) * DDIM + d] = acc[mi][ni][r];
        }
    }
}

// -------------- deterministic column mean of rate (two passes) ---------------
__global__ void k_colsum_part(const bf16* __restrict__ rt, float* __restrict__ part) {
    const int f = blockIdx.x * 256 + threadIdx.x;
    const int c = blockIdx.y;
    const int r0 = c * (NTOK / 16);
    float s = 0.f;
    for (int n = 0; n < NTOK / 16; ++n)
        s += (float)rt[(size_t)(r0 + n) * FDIM + f];
    part[(size_t)c * FDIM + f] = s;
}

__global__ void k_colsum_fin(const float* __restrict__ part, float* __restrict__ o) {
    const int f = blockIdx.x * 256 + threadIdx.x;
    float s = 0.f;
    #pragma unroll
    for (int c = 0; c < 16; ++c) s += part[(size_t)c * FDIM + f];
    o[f] = s * (1.0f / (float)NTOK);
}

// ------------------------------- launcher ------------------------------------
extern "C" void kernel_launch(void* const* d_in, const int* in_sizes, int n_in,
                              void* d_out, int out_size, void* d_ws, size_t ws_size,
                              hipStream_t stream) {
    (void)in_sizes; (void)n_in; (void)out_size; (void)ws_size;
    const float* x    = (const float*)d_in[0];
    const float* wup  = (const float*)d_in[1];
    const float* wdn  = (const float*)d_in[2];
    const float* beta = (const float*)d_in[3];
    const int*   Tp   = (const int*)d_in[4];

    char*  ws = (char*)d_ws;
    size_t o  = 0;
    auto alloc = [&](size_t bytes) -> void* {
        void* p = ws + o;
        o += (bytes + 255) & ~(size_t)255;
        return p;
    };
    bf16* xh   = (bf16*)alloc((size_t)NTOK * DDIM * 2);
    bf16* xl   = (bf16*)alloc((size_t)NTOK * DDIM * 2);
    bf16* wuh  = (bf16*)alloc((size_t)FDIM * DDIM * 2);
    bf16* wul  = (bf16*)alloc((size_t)FDIM * DDIM * 2);
    bf16* wdh  = (bf16*)alloc((size_t)DDIM * FDIM * 2);
    bf16* wdl  = (bf16*)alloc((size_t)DDIM * FDIM * 2);
    bf16* rate = (bf16*)alloc((size_t)NTOK * FDIM * 2);
    float* part = (float*)alloc((size_t)16 * FDIM * 4);

    const int nX = NTOK * DDIM;   // == FDIM*DDIM == DDIM*FDIM == 4M
    k_split<<<nX / 256, 256, 0, stream>>>(x,   xh,  xl,  nX);
    k_split<<<nX / 256, 256, 0, stream>>>(wup, wuh, wul, nX);
    k_split<<<nX / 256, 256, 0, stream>>>(wdn, wdh, wdl, nX);

    k_gemm1_lif<<<dim3(FDIM / BN, NTOK / BM), 256, 0, stream>>>(
        xh, xl, wuh, wul, beta, Tp, rate);

    k_colsum_part<<<dim3(FDIM / 256, 16), 256, 0, stream>>>(rate, part);
    k_colsum_fin<<<FDIM / 256, 256, 0, stream>>>(
        part, (float*)d_out + (size_t)NTOK * DDIM);

    k_gemm2<<<dim3(DDIM / BN, NTOK / BM), 256, 0, stream>>>(
        rate, wdh, wdl, (float*)d_out);
}
